// GRUDetector_37452114821595
// MI455X (gfx1250) — matrix operand order repaired and checked by reading, not compiled, not run on hardware
//
#include <hip/hip_runtime.h>
#include <hip/hip_bf16.h>

typedef __attribute__((ext_vector_type(16))) __bf16 v16bf;
typedef __attribute__((ext_vector_type(8)))  float  v8f;
typedef __attribute__((ext_vector_type(4)))  float  v4f;

#define B_ROWS 16      // batch rows per block
#define T_STEPS 512
#define F_IN 64
#define H_DIM 128
#define LDH (H_DIM + 8)   // bf16 h buffer row pitch
#define LDX (F_IN + 8)
#define LOG2E 1.4426950408889634f

// ---- fast transcendentals (single HW instructions, no IEEE div ladder) ----
__device__ __forceinline__ float fast_rcp(float x)  { return __builtin_amdgcn_rcpf(x); }
__device__ __forceinline__ float fast_exp2(float x) { return __builtin_amdgcn_exp2f(x); }
__device__ __forceinline__ float fast_sigmoid(float x) {
    return fast_rcp(1.0f + fast_exp2(-LOG2E * x));   // v_exp_f32 + v_rcp_f32
}
__device__ __forceinline__ float fast_tanh(float x) {
#if __has_builtin(__builtin_amdgcn_tanhf)
    return __builtin_amdgcn_tanhf(x);                // v_tanh_f32 (gfx1250)
#else
    return 1.0f - 2.0f * fast_rcp(1.0f + fast_exp2(2.0f * LOG2E * x));
#endif
}

__device__ __forceinline__ v8f wmma_bf16(v16bf a, v16bf b, v8f c) {
    // (neg_a, A, neg_b, B, c_mod, C, reuse_a, reuse_b)
    return __builtin_amdgcn_wmma_f32_16x16x32_bf16(false, a, false, b, (short)0, c, false, false);
}

// B fragment (32x16, KxN). Element (k,n) = W[n*ldw + k] (W row-major [N][K]).
// ISA layout: lanes0-15: N=lane, V0..3 = K0..7, V4..7 = K16..23; lanes16-31: N=lane-16, K+8.
// => element i<8 holds K = kb*32 + ((lane>=16)?8:0) + i; element 8+i holds that K + 16.
__device__ __forceinline__ v16bf load_bfrag_global(const float* __restrict__ W, int ldw,
                                                   int n0, int kb, int lane) {
    const int n  = n0 + (lane & 15);
    const int k0 = kb * 32 + ((lane >> 4) << 3);
    const float* p = W + (size_t)n * ldw + k0;
    v16bf f;
#pragma unroll
    for (int i = 0; i < 16; ++i) f[i] = (__bf16)p[(i & 7) + ((i >> 3) << 4)];
    return f;
}

// A fragment (16x32, MxK) from LDS bf16 [row][ld].
// ISA layout: lanes0-15 row=lane: V0..3 = K0..7, V4..7 = K16..23; lanes16-31: +8.
__device__ __forceinline__ v16bf load_afrag_lds(const __bf16* S, int ld, int kb, int lane) {
    const int row = lane & 15;
    const int hi8 = (lane >> 4) << 3;
    const __bf16* p = S + row * ld + kb * 32 + hi8;
    v16bf f;
#pragma unroll
    for (int i = 0; i < 8; ++i) f[i] = p[i];
#pragma unroll
    for (int i = 0; i < 8; ++i) f[8 + i] = p[16 + i];
    return f;
}

// One GRU timestep: read h/x from ping buffers, write h to pong buffer,
// stage x(t+1) into pong x-buffer, refill register prefetch with x(t+2).
// Exactly ONE barrier per step (ping-pong removes the write-after-read hazard).
__device__ __forceinline__ void gru_step(
    int t, const __bf16* hRead, __bf16* hWrite,
    const __bf16* xRead, __bf16* xWrite,
    const v16bf (&Bh)[3][4], const v16bf (&Bx)[3][2],
    v4f& xpre, float (&hstate)[8],
    const v4f* xsrc, int lane, int nc, int xr, int xf,
    float br, float bz, float bin, float bhn)
{
    // A fragments (h from previous step, x staged previous step)
    v16bf Ah[4];
#pragma unroll
    for (int kb = 0; kb < 4; ++kb) Ah[kb] = load_afrag_lds(hRead, LDH, kb, lane);
    v16bf Ax[2];
#pragma unroll
    for (int kb = 0; kb < 2; ++kb) Ax[kb] = load_afrag_lds(xRead, LDX, kb, lane);

    v8f accR = {}, accZ = {}, accXN = {}, accHN = {};
#pragma unroll
    for (int kb = 0; kb < 4; ++kb) {
        accR  = wmma_bf16(Ah[kb], Bh[0][kb], accR);
        accZ  = wmma_bf16(Ah[kb], Bh[1][kb], accZ);
        accHN = wmma_bf16(Ah[kb], Bh[2][kb], accHN);
    }
#pragma unroll
    for (int kb = 0; kb < 2; ++kb) {
        accR  = wmma_bf16(Ax[kb], Bx[0][kb], accR);
        accZ  = wmma_bf16(Ax[kb], Bx[1][kb], accZ);
        accXN = wmma_bf16(Ax[kb], Bx[2][kb], accXN);
    }

    // stage x(t+1) into the pong buffer, then refill prefetch with x(t+2):
    // the nontemporal load has (gates + barrier + next step's WMMAs) of slack.
    if (t + 1 < T_STEPS) {
        __bf16* xw = xWrite + xr * LDX + xf;
        xw[0] = (__bf16)xpre.x;  xw[1] = (__bf16)xpre.y;
        xw[2] = (__bf16)xpre.z;  xw[3] = (__bf16)xpre.w;
    }
    if (t + 2 < T_STEPS)
        xpre = __builtin_nontemporal_load(xsrc + (size_t)(t + 2) * (F_IN / 4));

    // gate math + state update. C/D layout: VGPR j -> row j + (lane>=16 ? 8:0), col = nc.
    // f32 state is lane-private -> lives in registers, never touches LDS.
    const int rbase = (lane >> 4) << 3;
#pragma unroll
    for (int j = 0; j < 8; ++j) {
        const float r = fast_sigmoid(accR[j] + br);
        const float z = fast_sigmoid(accZ[j] + bz);
        const float n = fast_tanh(accXN[j] + bin + r * (accHN[j] + bhn));
        const float hnew = (1.0f - z) * n + z * hstate[j];
        hstate[j] = hnew;
        hWrite[(rbase + j) * LDH + nc] = (__bf16)hnew;
    }
    __syncthreads();   // h/x pong buffers complete before next step reads them
}

__global__ __launch_bounds__(256, 1)
void gru_fused_kernel(const float* __restrict__ x,
                      const float* __restrict__ W_ih, const float* __restrict__ W_hh,
                      const float* __restrict__ b_ih, const float* __restrict__ b_hh,
                      const float* __restrict__ W1,   const float* __restrict__ b1,
                      const float* __restrict__ W2,   const float* __restrict__ b2,
                      float* __restrict__ out)
{
    __shared__ __bf16 hbuf[2][B_ROWS][LDH];   // ping-pong bf16 h (WMMA A operand)
    __shared__ __bf16 xbuf[2][B_ROWS][LDX];   // ping-pong x_t tile, bf16
    __shared__ float  hfin[B_ROWS][H_DIM];    // final f32 h for the head
    __shared__ float  hid [B_ROWS][32];       // head hidden layer

    const int tid  = threadIdx.x;
    const int lane = tid & 31;
    const int wave = tid >> 5;             // 0..7
    const int b0   = blockIdx.x * B_ROWS;
    const int cw   = wave * 16;            // this wave's H-column tile (same col in r,z,n)

    // zero initial hidden state (ping buffer read at t=0)
    for (int i = tid; i < 2 * B_ROWS * LDH; i += 256) (&hbuf[0][0][0])[i] = (__bf16)0.0f;

    // per-lane gate biases (N = cw + lane&15 is fixed per lane in the C/D layout)
    const int nc = cw + (lane & 15);
    const float br  = b_ih[nc]        + b_hh[nc];
    const float bz  = b_ih[128 + nc]  + b_hh[128 + nc];
    const float bin = b_ih[256 + nc];
    const float bhn = b_hh[256 + nc];

    // Preload all weight B-fragments into VGPRs (loop-invariant: zero per-step weight traffic)
    v16bf Bh[3][4];   // [gate][kblock]  W_hh: K=128 -> 4 blocks of 32
    v16bf Bx[3][2];   // [gate][kblock]  W_ih: K=64  -> 2 blocks of 32
#pragma unroll
    for (int g = 0; g < 3; ++g) {
#pragma unroll
        for (int kb = 0; kb < 4; ++kb)
            Bh[g][kb] = load_bfrag_global(W_hh, H_DIM, g * 128 + cw, kb, lane);
#pragma unroll
        for (int kb = 0; kb < 2; ++kb)
            Bx[g][kb] = load_bfrag_global(W_ih, F_IN, g * 128 + cw, kb, lane);
    }

    // x staging: thread's 4 elements are one contiguous v4f on a fixed row,
    // address advances by F_IN floats per step. Non-temporal: x is streamed once.
    const int idx0 = tid * 4;
    const int xr   = idx0 >> 6;            // 0..15
    const int xf   = idx0 & 63;            // multiple of 4
    const v4f* xsrc = (const v4f*)(x + (size_t)(b0 + xr) * T_STEPS * F_IN + xf);

    // stage x(0) directly, prefetch x(1) into registers
    {
        v4f x0 = __builtin_nontemporal_load(xsrc);
        __bf16* xw = &xbuf[0][xr][xf];
        xw[0] = (__bf16)x0.x;  xw[1] = (__bf16)x0.y;
        xw[2] = (__bf16)x0.z;  xw[3] = (__bf16)x0.w;
    }
    v4f xpre = __builtin_nontemporal_load(xsrc + (F_IN / 4));

    float hstate[8];
#pragma unroll
    for (int j = 0; j < 8; ++j) hstate[j] = 0.0f;

    __syncthreads();

    for (int t = 0; t < T_STEPS; t += 2) {
        gru_step(t,     &hbuf[0][0][0], &hbuf[1][0][0], &xbuf[0][0][0], &xbuf[1][0][0],
                 Bh, Bx, xpre, hstate, xsrc, lane, nc, xr, xf, br, bz, bin, bhn);
        gru_step(t + 1, &hbuf[1][0][0], &hbuf[0][0][0], &xbuf[1][0][0], &xbuf[0][0][0],
                 Bh, Bx, xpre, hstate, xsrc, lane, nc, xr, xf, br, bz, bin, bhn);
    }

    // dump lane-private f32 state for the head
    {
        const int rbase = (lane >> 4) << 3;
#pragma unroll
        for (int j = 0; j < 8; ++j) hfin[rbase + j][nc] = hstate[j];
    }
    __syncthreads();

    // --- head: hidden = relu(hT @ W1^T + b1); out = hidden @ W2^T + b2 ---
    for (int idx = tid; idx < B_ROWS * 32; idx += 256) {
        const int r = idx >> 5, c = idx & 31;
        float s = b1[c];
        const float* w = W1 + c * H_DIM;
#pragma unroll 8
        for (int k = 0; k < H_DIM; ++k) s += hfin[r][k] * w[k];
        hid[r][c] = fmaxf(s, 0.0f);
    }
    __syncthreads();
    if (tid < B_ROWS) {
        float s = b2[0];
#pragma unroll
        for (int c = 0; c < 32; ++c) s += hid[tid][c] * W2[c];
        out[b0 + tid] = s;
    }
}

extern "C" void kernel_launch(void* const* d_in, const int* in_sizes, int n_in,
                              void* d_out, int out_size, void* d_ws, size_t ws_size,
                              hipStream_t stream) {
    (void)in_sizes; (void)n_in; (void)d_ws; (void)ws_size; (void)out_size;
    const float* x    = (const float*)d_in[0];
    const float* W_ih = (const float*)d_in[1];
    const float* W_hh = (const float*)d_in[2];
    const float* b_ih = (const float*)d_in[3];
    const float* b_hh = (const float*)d_in[4];
    const float* W1   = (const float*)d_in[5];
    const float* b1   = (const float*)d_in[6];
    const float* W2   = (const float*)d_in[7];
    const float* b2   = (const float*)d_in[8];
    float* out = (float*)d_out;

    gru_fused_kernel<<<1024 / B_ROWS, 256, 0, stream>>>(
        x, W_ih, W_hh, b_ih, b_hh, W1, b1, W2, b2, out);
}
